// NLPAttention_20246475834055
// MI455X (gfx1250) — matrix operand order repaired and checked by reading, not compiled, not run on hardware
//
#include <hip/hip_runtime.h>

typedef __attribute__((ext_vector_type(16))) __bf16 v16bf;
typedef __attribute__((ext_vector_type(8)))  __bf16 v8bf;
typedef __attribute__((ext_vector_type(4)))  __bf16 v4bf;
typedef __attribute__((ext_vector_type(8)))  float  v8f;
typedef __attribute__((ext_vector_type(4)))  float  vf4;

#define B_    4
#define H_    8
#define S_    2048
#define D_    64
#define BLK_M 128             // 4 waves x 2 query tiles of 16
#define BLK_N 32
#define VT_STRIDE 40          // 32 + 8 pad: rows 80B -> 16B-aligned, banks spread
#define LOG2E     1.44269504088896340736f
#define NEG_L2    (-1.4426950408889634e9f)   // -1e9 * log2e (log2-domain bias)

static __device__ __forceinline__ v16bf cat8(v8bf a, v8bf b) {
    return __builtin_shufflevector(a, b, 0, 1, 2, 3, 4, 5, 6, 7,
                                         8, 9, 10, 11, 12, 13, 14, 15);
}

static __device__ __forceinline__ v8bf pack8(v8f x) {
    v8bf r;
    #pragma unroll
    for (int j = 0; j < 8; ++j) r[j] = (__bf16)x[j];   // pairs -> v_cvt_pk_bf16_f32
    return r;
}

static __device__ __forceinline__ v8bf xor16_v8bf(v8bf v) {
    int w[4];
    __builtin_memcpy(w, &v, 16);
    #pragma unroll
    for (int j = 0; j < 4; ++j) w[j] = __shfl_xor(w[j], 16, 32);
    v8bf r;
    __builtin_memcpy(&r, w, 16);
    return r;
}

__global__ __launch_bounds__(128) void NLPAttention_fa_kernel(
    const float* __restrict__ qs, const float* __restrict__ ks,
    const float* __restrict__ vs, const float* __restrict__ mask,
    float* __restrict__ out)
{
    __shared__ __bf16 Kbf[BLK_N][D_];           // K block, row-major bf16 (4KB)
    __shared__ __bf16 Vt[D_][VT_STRIDE];        // V block, transposed bf16 (5KB)

    const int tid  = threadIdx.x;
    const int lane = tid & 31;
    const int wave = tid >> 5;
    const int hl   = lane & 15;
    const bool lo  = lane < 16;

    const int bh    = blockIdx.y;
    const int b     = bh >> 3;                  // H_ == 8
    const int qbase = blockIdx.x * BLK_M + wave * 32;   // 2 tiles per wave

    const size_t qkv_off = (size_t)bh * S_ * D_;
    const float* Q = qs + qkv_off;
    const float* K = ks + qkv_off;
    const float* V = vs + qkv_off;
    const float* M = mask + (size_t)b * S_ * S_;

    // K staging: 32 rows x 64 floats / 128 threads = 16 contiguous floats each
    const int sr = tid >> 2;                    // block-local row 0..31
    const int sd = (tid & 3) * 16;              // d base 0,16,32,48
    // V staging: 4x4 sub-tile per thread, transposed in registers
    const int vrg = (tid >> 4) * 4;             // row group base 0,4,..,28
    const int vcg = (tid & 15) * 4;             // col group base 0,4,..,60

    const int abase = lo ? 0 : 8;               // accumulator row base per half (8h)
    const int dlo   = lo ? 0 : 16;              // B-layout K-dim split per half

    int qrow[2];                                // this lane's query per tile
    qrow[0] = qbase + hl;
    qrow[1] = qbase + 16 + hl;

    // ---- Q B-fragments (32d x 16q) x 2 tiles; fold 1/sqrt(d)*log2e in ----
    v16bf qb[2][2];
    #pragma unroll
    for (int u = 0; u < 2; ++u)
        #pragma unroll
        for (int c = 0; c < 2; ++c) {
            const float* qp = Q + (size_t)qrow[u] * D_ + c * 32 + dlo;
            #pragma unroll
            for (int h = 0; h < 16; ++h)
                qb[u][c][h] = (__bf16)(qp[h] * (0.125f * LOG2E));
        }

    v8f o[2][4];                                // O^T accumulators (d x q) x 2
    #pragma unroll
    for (int u = 0; u < 2; ++u)
        #pragma unroll
        for (int t = 0; t < 4; ++t) o[u][t] = (v8f){};
    float mq[2] = { -__builtin_inff(), -__builtin_inff() };
    float lq[2] = { 0.f, 0.f };

    for (int kb = 0; kb < S_; kb += BLK_N) {
        if (kb + BLK_N < S_) {                  // gfx1250 global_prefetch_b8
            __builtin_prefetch(K + (size_t)(kb + BLK_N) * D_ + tid * 16, 0, 3);
            __builtin_prefetch(V + (size_t)(kb + BLK_N) * D_ + tid * 16, 0, 3);
        }

        __syncthreads();                        // prior block's LDS reads done

        // ---- cooperative stage: K -> Kbf (row major) ----
        {
            const float* ksrc = K + (size_t)(kb + sr) * D_ + sd;
            vf4 a0 = *(const vf4*)(ksrc + 0);
            vf4 a1 = *(const vf4*)(ksrc + 4);
            vf4 a2 = *(const vf4*)(ksrc + 8);
            vf4 a3 = *(const vf4*)(ksrc + 12);
            v8bf klo, khi;
            #pragma unroll
            for (int j = 0; j < 4; ++j) {
                klo[j]     = (__bf16)a0[j];
                klo[j + 4] = (__bf16)a1[j];
                khi[j]     = (__bf16)a2[j];
                khi[j + 4] = (__bf16)a3[j];
            }
            *(v8bf*)&Kbf[sr][sd]     = klo;
            *(v8bf*)&Kbf[sr][sd + 8] = khi;
        }
        // ---- V -> Vt: 4x4 register transpose, packed ds_store_b64 ----
        {
            vf4 vr0 = *(const vf4*)(V + (size_t)(kb + vrg + 0) * D_ + vcg);
            vf4 vr1 = *(const vf4*)(V + (size_t)(kb + vrg + 1) * D_ + vcg);
            vf4 vr2 = *(const vf4*)(V + (size_t)(kb + vrg + 2) * D_ + vcg);
            vf4 vr3 = *(const vf4*)(V + (size_t)(kb + vrg + 3) * D_ + vcg);
            #pragma unroll
            for (int j = 0; j < 4; ++j) {
                v4bf col;
                col[0] = (__bf16)vr0[j];
                col[1] = (__bf16)vr1[j];
                col[2] = (__bf16)vr2[j];
                col[3] = (__bf16)vr3[j];
                *(v4bf*)&Vt[vcg + j][vrg] = col;   // 8B store, 8B-aligned
            }
        }

        __syncthreads();

        // ---- S^T = K Q^T : K A-frags loaded once, reused for both q tiles ----
        v8f s[2][2];
        #pragma unroll
        for (int u = 0; u < 2; ++u) { s[u][0] = (v8f){}; s[u][1] = (v8f){}; }
        #pragma unroll
        for (int c = 0; c < 2; ++c) {
            v16bf ka0 = cat8(*(const v8bf*)&Kbf[hl     ][c * 32 + dlo],
                             *(const v8bf*)&Kbf[hl     ][c * 32 + dlo + 8]);
            v16bf ka1 = cat8(*(const v8bf*)&Kbf[16 + hl][c * 32 + dlo],
                             *(const v8bf*)&Kbf[16 + hl][c * 32 + dlo + 8]);
            #pragma unroll
            for (int u = 0; u < 2; ++u) {
                s[u][0] = __builtin_amdgcn_wmma_f32_16x16x32_bf16(false, ka0, false, qb[u][c], (short)0, s[u][0], false, false);
                s[u][1] = __builtin_amdgcn_wmma_f32_16x16x32_bf16(false, ka1, false, qb[u][c], (short)0, s[u][1], false, false);
            }
        }

        // ---- per-tile mask + online softmax + P^T fragment ----
        v16bf pb[2];
        #pragma unroll
        for (int u = 0; u < 2; ++u) {
            const float* mp = M + (size_t)qrow[u] * S_ + kb + abase;
            vf4 m0a = *(const vf4*)(mp + 0);
            vf4 m0b = *(const vf4*)(mp + 4);
            vf4 m1a = *(const vf4*)(mp + 16);
            vf4 m1b = *(const vf4*)(mp + 20);
            float mk0[8], mk1[8];
            #pragma unroll
            for (int j = 0; j < 4; ++j) {
                mk0[j] = m0a[j]; mk0[j + 4] = m0b[j];
                mk1[j] = m1a[j]; mk1[j + 4] = m1b[j];
            }
            v8f s0 = s[u][0], s1 = s[u][1];
            #pragma unroll
            for (int r = 0; r < 8; ++r) {       // log2-domain: s*mk + (1-mk)*NEG*log2e
                float t0 = __builtin_fmaf(mk0[r], -NEG_L2, NEG_L2);
                float t1 = __builtin_fmaf(mk1[r], -NEG_L2, NEG_L2);
                s0[r] = __builtin_fmaf(s0[r], mk0[r], t0);
                s1[r] = __builtin_fmaf(s1[r], mk1[r], t1);
            }

            float mx01 = fmaxf(fmaxf(s0[0], s1[0]), fmaxf(s0[1], s1[1]));
            float mx23 = fmaxf(fmaxf(s0[2], s1[2]), fmaxf(s0[3], s1[3]));
            float mx45 = fmaxf(fmaxf(s0[4], s1[4]), fmaxf(s0[5], s1[5]));
            float mx67 = fmaxf(fmaxf(s0[6], s1[6]), fmaxf(s0[7], s1[7]));
            float mx = fmaxf(fmaxf(mx01, mx23), fmaxf(mx45, mx67));
            mx = fmaxf(mx, __shfl_xor(mx, 16, 32));
            float mnew = fmaxf(mq[u], mx);
            float corr = __builtin_amdgcn_exp2f(mq[u] - mnew);   // bare v_exp_f32
            mq[u] = mnew;
            #pragma unroll
            for (int r = 0; r < 8; ++r) {
                s0[r] = __builtin_amdgcn_exp2f(s0[r] - mnew);
                s1[r] = __builtin_amdgcn_exp2f(s1[r] - mnew);
            }
            float ps = 0.f;
            #pragma unroll
            for (int r = 0; r < 8; ++r) ps += s0[r] + s1[r];
            ps += __shfl_xor(ps, 16, 32);
            lq[u] = lq[u] * corr + ps;          // denominator keeps masked ~0 terms
            #pragma unroll
            for (int t = 0; t < 4; ++t)
                #pragma unroll
                for (int r = 0; r < 8; ++r) o[u][t][r] *= corr;
            #pragma unroll
            for (int r = 0; r < 8; ++r) {       // multiplicative post-mask
                s0[r] *= mk0[r];
                s1[r] *= mk1[r];
            }

            // P^T B-fragment: pack + one half-wave exchange
            v8bf p0 = pack8(s0);                // lo: keys 0-7   | hi: keys 8-15
            v8bf p1 = pack8(s1);                // lo: keys 16-23 | hi: keys 24-31
            v8bf send = lo ? p1 : p0;
            v8bf recv = xor16_v8bf(send);
            v8bf first8  = lo ? p0   : recv;    // lo: keys 0-7   | hi: keys 16-23
            v8bf second8 = lo ? recv : p1;      // lo: keys 8-15  | hi: keys 24-31
            pb[u] = cat8(first8, second8);
        }

        // ---- O^T += V^T P^T : V A-frags loaded once, reused for both tiles ----
        #pragma unroll
        for (int t = 0; t < 4; ++t) {
            v16bf va = cat8(*(const v8bf*)&Vt[t * 16 + hl][dlo],
                            *(const v8bf*)&Vt[t * 16 + hl][dlo + 8]);
            #pragma unroll
            for (int u = 0; u < 2; ++u)
                o[u][t] = __builtin_amdgcn_wmma_f32_16x16x32_bf16(false, va, false, pb[u], (short)0, o[u][t], false, false);
        }
    }

    // ---- normalize + store: per-lane contiguous 8-float runs ----
    #pragma unroll
    for (int u = 0; u < 2; ++u) {
        const float linv = 1.0f / lq[u];
        #pragma unroll
        for (int t = 0; t < 4; ++t) {
            vf4 w0, w1;
            #pragma unroll
            for (int j = 0; j < 4; ++j) {
                w0[j] = o[u][t][j]     * linv;
                w1[j] = o[u][t][j + 4] * linv;
            }
            float* dst = out + qkv_off + (size_t)qrow[u] * D_ + t * 16 + abase;
            *(vf4*)(dst)     = w0;
            *(vf4*)(dst + 4) = w1;
        }
    }
}

extern "C" void kernel_launch(void* const* d_in, const int* in_sizes, int n_in,
                              void* d_out, int out_size, void* d_ws, size_t ws_size,
                              hipStream_t stream) {
    (void)in_sizes; (void)n_in; (void)d_ws; (void)ws_size; (void)out_size;
    const float* qs   = (const float*)d_in[0];
    const float* ks   = (const float*)d_in[1];
    const float* vs   = (const float*)d_in[2];
    const float* mask = (const float*)d_in[3];
    float* out = (float*)d_out;

    dim3 grid(S_ / BLK_M, B_ * H_);   // 16 x 32 workgroups
    dim3 block(128);                  // 4 wave32 waves, 32 queries per wave
    NLPAttention_fa_kernel<<<grid, block, 0, stream>>>(qs, ks, vs, mask, out);
}
